// AttentionLayer2_5360119185643
// MI455X (gfx1250) — matrix operand order repaired and checked by reading, not compiled, not run on hardware
//
#include <hip/hip_runtime.h>
#include <hip/hip_bf16.h>
#include <stdint.h>

// ---------------- problem constants (match reference) ----------------
#define HH    8      // heads
#define HD    16     // head dim
#define DD    128    // feature dim
#define HID   512    // hidden dim
#define SLOPE 0.2f   // LeakyReLU negative slope
#define EPS   1e-5f

// ---------------- CDNA5 WMMA vector types ----------------
typedef __attribute__((ext_vector_type(16))) __bf16         v16bf;
typedef __attribute__((ext_vector_type(8)))  __bf16         v8bf;
typedef __attribute__((ext_vector_type(8)))  float          v8f;
typedef __attribute__((ext_vector_type(8)))  unsigned short v8u;
typedef __attribute__((ext_vector_type(4)))  unsigned short v4u;
// exact pointee type expected by the async-to-LDS builtin (gcc-style int4)
typedef __attribute__((__vector_size__(16))) int            v4i_b128;

// use the CDNA5 async global->LDS path (double buffered) in the GEMM
#define USE_ASYNC_LDS 1

__device__ __forceinline__ unsigned short f2bf(float f) {
  unsigned u = __float_as_uint(f);
  u += 0x7FFFu + ((u >> 16) & 1u);         // round-to-nearest-even
  return (unsigned short)(u >> 16);
}

// order-preserving monotone encoding of float into u32 (for atomicMax)
__device__ __forceinline__ unsigned fenc(float f) {
  unsigned u = __float_as_uint(f);
  return (u & 0x80000000u) ? ~u : (u | 0x80000000u);
}
__device__ __forceinline__ float fdec(unsigned u) {
  return __uint_as_float((u & 0x80000000u) ? (u ^ 0x80000000u) : ~u);
}

// ---- async global->LDS 16B copy (GLOBAL_LOAD_ASYNC_TO_LDS_B128) ----
__device__ __forceinline__ void async_cp16(const unsigned short* g,
                                           unsigned short* l) {
#if USE_ASYNC_LDS
#if __has_builtin(__builtin_amdgcn_global_load_async_to_lds_b128)
  __builtin_amdgcn_global_load_async_to_lds_b128(
      (__attribute__((address_space(1))) v4i_b128*)(void*)g,
      (__attribute__((address_space(3))) v4i_b128*)(void*)l, 0, 0);
#else
  unsigned laddr =
      (unsigned)(unsigned long long)(__attribute__((address_space(3))) void*)l;
  asm volatile("global_load_async_to_lds_b128 %0, %1, off"
               :: "v"(laddr), "v"(g) : "memory");
#endif
#else
  *(v8u*)l = *(const v8u*)g;
#endif
}

__device__ __forceinline__ void wait_async() {
#if USE_ASYNC_LDS
#if __has_builtin(__builtin_amdgcn_s_wait_asynccnt)
  __builtin_amdgcn_s_wait_asynccnt(0);
#else
  asm volatile("s_wait_asynccnt 0" ::: "memory");
#endif
#endif
}

// ---------------- utility kernels ----------------
__global__ void zero_f32(float* p, long long n) {
  long long i = (long long)blockIdx.x * blockDim.x + threadIdx.x;
  long long s = (long long)gridDim.x * blockDim.x;
  for (; i < n; i += s) p[i] = 0.f;
}

// f32 -> bf16, 4 elements/thread (n4 = count/4)
__global__ void cvt_bf16_v4(const float* __restrict__ in,
                            unsigned short* __restrict__ out, long long n4) {
  long long i = (long long)blockIdx.x * blockDim.x + threadIdx.x;
  long long s = (long long)gridDim.x * blockDim.x;
  for (; i < n4; i += s) {
    float4 v = ((const float4*)in)[i];
    v4u o;
    o[0] = f2bf(v.x); o[1] = f2bf(v.y); o[2] = f2bf(v.z); o[3] = f2bf(v.w);
    ((v4u*)out)[i] = o;
  }
}

// f32 [K][Nc] -> bf16 transposed [Nc][K]  (weights only: tiny, done once)
__global__ void cvt_bf16_t(const float* __restrict__ in,
                           unsigned short* __restrict__ out, int K, int Nc) {
  long long i = (long long)blockIdx.x * blockDim.x + threadIdx.x;
  if (i >= (long long)K * Nc) return;
  int n = (int)(i / K), k = (int)(i % K);
  out[i] = f2bf(in[(size_t)k * Nc + n]);
}

// ---------------- WMMA bf16 GEMM: C[M,Nc] = A[M,K] @ B[K,Nc] ----------------
// A is [M][K] row-major bf16; B is passed PRE-TRANSPOSED as Bt[Nc][K] so both
// LDS tiles are staged with straight 16B copies (async global->LDS, double
// buffered: prefetch K-step s+1 while WMMAs run on step s).
// Block: 128 threads (4 waves), tile 64x64, K-step 32.
// EPI: 0 = f32 out; 1 = +bias, ReLU, bf16 out; 2 = +bias, f32 out.
template <int EPI>
__global__ __launch_bounds__(128)
void wmma_gemm(const unsigned short* __restrict__ A,
               const unsigned short* __restrict__ Bt,
               void* __restrict__ Cout,
               const float* __restrict__ bias,
               int M, int K, int Ncols) {
  // row stride 40 elems (80B): keeps 16B alignment, skews LDS banks
  __shared__ __align__(16) unsigned short lA[2][64][40];   // [buf][m][k]
  __shared__ __align__(16) unsigned short lB[2][64][40];   // [buf][n][k]

  const int tid  = threadIdx.x;
  const int wave = tid >> 5;
  const int lane = tid & 31;
  const int lr   = lane & 15;     // row (A) / col (B/C) within 16
  const int hi   = lane >> 4;     // lane-half selector
  const int rowBase = blockIdx.y * 64;
  const int colBase = blockIdx.x * 64;

  // this thread's two 16B chunks per tile: chunk cid -> (row cid>>2, k (cid&3)*8)
  const int r0 = tid >> 2,         k0c = (tid & 3) * 8;         // cid = tid
  const int r1 = (tid + 128) >> 2, k1c = ((tid + 128) & 3) * 8; // cid = tid+128

  v8f acc[4];
#pragma unroll
  for (int c = 0; c < 4; ++c)
#pragma unroll
    for (int r = 0; r < 8; ++r) acc[c][r] = 0.f;

  auto stage = [&](int k0, int buf) {
    // A tile 64x32 (rows >= M left stale; their outputs are never stored)
    if (rowBase + r0 < M)
      async_cp16(A + (size_t)(rowBase + r0) * K + k0 + k0c, &lA[buf][r0][k0c]);
    if (rowBase + r1 < M)
      async_cp16(A + (size_t)(rowBase + r1) * K + k0 + k1c, &lA[buf][r1][k1c]);
    // B tile 64(n) x 32(k) from pre-transposed Bt (always in range)
    async_cp16(Bt + (size_t)(colBase + r0) * K + k0 + k0c, &lB[buf][r0][k0c]);
    async_cp16(Bt + (size_t)(colBase + r1) * K + k0 + k1c, &lB[buf][r1][k1c]);
  };

  const int nsteps = K >> 5;
  stage(0, 0);

  for (int s = 0; s < nsteps; ++s) {
    const int buf = s & 1;
    wait_async();       // own async copies for step s complete
    __syncthreads();    // everyone's complete; also fences reads of buf^1
    if (s + 1 < nsteps) stage((s + 1) << 5, buf ^ 1);

    // A fragment (ISA 16-bit A 16x32 layout): lane row = lr,
    // elems 0..7 -> K = 8*hi + j, elems 8..15 -> K = 16 + 8*hi + j
    union { v8bf b; v8u u; } a0, a1;
    a0.u = *(const v8u*)&lA[buf][16 * wave + lr][hi * 8];
    a1.u = *(const v8u*)&lA[buf][16 * wave + lr][16 + hi * 8];
    v16bf afrag;
#pragma unroll
    for (int j = 0; j < 8; ++j) { afrag[j] = a0.b[j]; afrag[8 + j] = a1.b[j]; }

#pragma unroll
    for (int c = 0; c < 4; ++c) {
      // B fragment: col = lr, elems 0..7 -> K = 8*hi + j, elems 8..15 -> K = 16 + 8*hi + j
      union { v8bf b; v8u u; } b0, b1;
      b0.u = *(const v8u*)&lB[buf][16 * c + lr][hi * 8];
      b1.u = *(const v8u*)&lB[buf][16 * c + lr][16 + hi * 8];
      v16bf bfrag;
#pragma unroll
      for (int j = 0; j < 8; ++j) { bfrag[j] = b0.b[j]; bfrag[8 + j] = b1.b[j]; }
      acc[c] = __builtin_amdgcn_wmma_f32_16x16x32_bf16(
          false, afrag, false, bfrag, (short)0, acc[c], false, false);
    }
  }

  // epilogue (C layout: VGPR r -> M = 16w + 8*hi + r, N = 16c + lr)
#pragma unroll
  for (int c = 0; c < 4; ++c) {
    int col = colBase + 16 * c + lr;
#pragma unroll
    for (int r = 0; r < 8; ++r) {
      int row = rowBase + 16 * wave + 8 * hi + r;
      if (row < M) {
        float v = acc[c][r];
        if (EPI == 0) {
          ((float*)Cout)[(size_t)row * Ncols + col] = v;
        } else if (EPI == 1) {
          v += bias[col];
          v = v > 0.f ? v : 0.f;
          ((unsigned short*)Cout)[(size_t)row * Ncols + col] = f2bf(v);
        } else {
          ((float*)Cout)[(size_t)row * Ncols + col] = v + bias[col];
        }
      }
    }
  }
}

// ---------------- per-node attention logits el/er ----------------
__global__ void node_logits_k(const float* __restrict__ feat,
                              const float* __restrict__ attn_l,
                              const float* __restrict__ attn_r,
                              float* __restrict__ el, float* __restrict__ er,
                              long long NHt) {
  long long i = (long long)blockIdx.x * blockDim.x + threadIdx.x;
  if (i >= NHt) return;
  long long n = i >> 3;
  int h = (int)(i & 7);
  const float* f  = feat + (size_t)n * DD + h * HD;
  const float* al = attn_l + h * HD;
  const float* ar = attn_r + h * HD;
  float sl = 0.f, sr = 0.f;
#pragma unroll
  for (int j = 0; j < HD; ++j) { float x = f[j]; sl += x * al[j]; sr += x * ar[j]; }
  el[i] = sl;
  er[i] = sr;
}

__device__ __forceinline__ float edge_e(const int* src, const int* dst,
                                        const float* el, const float* er,
                                        long long i, int& s, int& d, int& h) {
  long long e = i >> 3;
  h = (int)(i & 7);
  s = src[e];
  d = dst[e];
  float v = el[(size_t)s * HH + h] + er[(size_t)d * HH + h];
  return v > 0.f ? v : SLOPE * v;
}

// pass 1: segment max via ordered-uint atomicMax (resolves in L2)
__global__ void edge_max_k(const int* __restrict__ src, const int* __restrict__ dst,
                           const float* __restrict__ el, const float* __restrict__ er,
                           unsigned* __restrict__ menc, long long EHt) {
  long long i = (long long)blockIdx.x * blockDim.x + threadIdx.x;
  if (i >= EHt) return;
  int s, d, h;
  float v = edge_e(src, dst, el, er, i, s, d, h);
  atomicMax(&menc[(size_t)d * HH + h], fenc(v));
}

// pass 2: segment sum of exp(e - m)
__global__ void edge_sum_k(const int* __restrict__ src, const int* __restrict__ dst,
                           const float* __restrict__ el, const float* __restrict__ er,
                           const unsigned* __restrict__ menc,
                           float* __restrict__ denom, long long EHt) {
  long long i = (long long)blockIdx.x * blockDim.x + threadIdx.x;
  if (i >= EHt) return;
  int s, d, h;
  float v = edge_e(src, dst, el, er, i, s, d, h);
  float m = fdec(menc[(size_t)d * HH + h]);
  atomicAdd(&denom[(size_t)d * HH + h], __expf(v - m));
}

// pass 3: alpha-weighted scatter of feat[src] into agg[dst] (agg fits in L2)
__global__ void edge_agg_k(const int* __restrict__ src, const int* __restrict__ dst,
                           const float* __restrict__ el, const float* __restrict__ er,
                           const unsigned* __restrict__ menc,
                           const float* __restrict__ denom,
                           const float* __restrict__ feat,
                           float* __restrict__ agg, long long EHt) {
  long long i = (long long)blockIdx.x * blockDim.x + threadIdx.x;
  if (i >= EHt) return;
  int s, d, h;
  float v = edge_e(src, dst, el, er, i, s, d, h);
  float m = fdec(menc[(size_t)d * HH + h]);
  float alpha = __expf(v - m) / denom[(size_t)d * HH + h];
  const float4* f = (const float4*)(feat + (size_t)s * DD + h * HD);
  float* o = agg + (size_t)d * DD + h * HD;
#pragma unroll
  for (int j = 0; j < 4; ++j) {
    float4 q = f[j];
    atomicAdd(o + 4 * j + 0, q.x * alpha);
    atomicAdd(o + 4 * j + 1, q.y * alpha);
    atomicAdd(o + 4 * j + 2, q.z * alpha);
    atomicAdd(o + 4 * j + 3, q.w * alpha);
  }
}

// ---------------- BatchNorm (training-mode, batch stats) ----------------
__global__ __launch_bounds__(128)
void bn_stats_k(const float* __restrict__ v, float* __restrict__ sum,
                float* __restrict__ sq, int M, int rowsPerBlk) {
  int col = threadIdx.x;                      // 128 columns
  int r0 = blockIdx.x * rowsPerBlk;
  int r1 = r0 + rowsPerBlk; if (r1 > M) r1 = M;
  float s = 0.f, q = 0.f;
  for (int r = r0; r < r1; ++r) {
    float x = v[(size_t)r * DD + col];
    s += x; q += x * x;
  }
  atomicAdd(&sum[col], s);
  atomicAdd(&sq[col], q);
}

template <bool BF16OUT>
__global__ void bn_apply_k(const float* __restrict__ v,
                           const float* __restrict__ sum, const float* __restrict__ sq,
                           const float* __restrict__ gamma, const float* __restrict__ beta,
                           void* __restrict__ out, int M) {
  long long i = (long long)blockIdx.x * blockDim.x + threadIdx.x;
  if (i >= (long long)M * DD) return;
  int col = (int)(i & (DD - 1));
  float inv = 1.f / (float)M;
  float mean = sum[col] * inv;
  float var  = sq[col] * inv - mean * mean;
  float y = (v[i] - mean) * rsqrtf(var + EPS) * gamma[col] + beta[col];
  if (BF16OUT) ((unsigned short*)out)[i] = f2bf(y);
  else         ((float*)out)[i] = y;
}

// ---------------- host launcher ----------------
extern "C" void kernel_launch(void* const* d_in, const int* in_sizes, int n_in,
                              void* d_out, int out_size, void* d_ws, size_t ws_size,
                              hipStream_t stream) {
  (void)n_in; (void)out_size; (void)ws_size;
  const float* x      = (const float*)d_in[0];
  const int*   src    = (const int*)  d_in[1];
  const int*   dst    = (const int*)  d_in[2];
  const float* W      = (const float*)d_in[3];
  const float* attn_l = (const float*)d_in[4];
  const float* attn_r = (const float*)d_in[5];
  // d_in[6] = bias_gat: per-column constant shift immediately followed by
  // training-mode BN -> cancels exactly (mean absorbs it). Intentionally unused.
  const float* bn1_g  = (const float*)d_in[7];
  const float* bn1_b  = (const float*)d_in[8];
  const float* W1     = (const float*)d_in[9];
  const float* b1     = (const float*)d_in[10];
  const float* W2     = (const float*)d_in[11];
  const float* b2     = (const float*)d_in[12];
  const float* bn2_g  = (const float*)d_in[13];
  const float* bn2_b  = (const float*)d_in[14];

  const long long Nn = in_sizes[0] / DD;
  const long long E  = in_sizes[1];
  const int Mi = (int)Nn;

  // ---- workspace carving (256B aligned) ----
  char* ws = (char*)d_ws;
  size_t off = 0;
  auto carve = [&](size_t bytes) -> char* {
    char* p = ws + off;
    off = (off + bytes + 255) & ~(size_t)255;
    return p;
  };
  // zero-initialized block first (contiguous so one fill kernel covers it)
  unsigned* menc  = (unsigned*)carve((size_t)Nn * HH * 4);
  float*    denom = (float*)   carve((size_t)Nn * HH * 4);
  float*    agg   = (float*)   carve((size_t)Nn * DD * 4);
  float*    sum1  = (float*)   carve(DD * 4);
  float*    sq1   = (float*)   carve(DD * 4);
  float*    sum2  = (float*)   carve(DD * 4);
  float*    sq2   = (float*)   carve(DD * 4);
  const long long zero_floats = (long long)(off / 4);
  // the rest (weights stored bf16 AND transposed: [N][K])
  unsigned short* x_bf   = (unsigned short*)carve((size_t)Nn * DD * 2);
  unsigned short* Wt_bf  = (unsigned short*)carve((size_t)DD * DD * 2);
  unsigned short* W1t_bf = (unsigned short*)carve((size_t)DD * HID * 2);
  unsigned short* W2t_bf = (unsigned short*)carve((size_t)HID * DD * 2);
  float*          feat   = (float*)         carve((size_t)Nn * DD * 4);
  float*          el     = (float*)         carve((size_t)Nn * HH * 4);
  float*          er     = (float*)         carve((size_t)Nn * HH * 4);
  unsigned short* hn_bf  = (unsigned short*)carve((size_t)Nn * DD * 2);
  unsigned short* act_bf = (unsigned short*)carve((size_t)Nn * HID * 2);
  float*          y      = feat;  // feat dead after edge_agg; reuse for GEMM3 out

  const dim3 blk(256);
  const dim3 gblk(128);

  // converts (weights transposed to [N][K]) + zero init
  long long nx = Nn * DD;
  cvt_bf16_v4<<<dim3((unsigned)((nx / 4 + 255) / 256)), blk, 0, stream>>>(x, x_bf, nx / 4);
  cvt_bf16_t<<<dim3((DD * DD + 255) / 256), blk, 0, stream>>>(W, Wt_bf, DD, DD);
  cvt_bf16_t<<<dim3((DD * HID + 255) / 256), blk, 0, stream>>>(W1, W1t_bf, DD, HID);
  cvt_bf16_t<<<dim3((HID * DD + 255) / 256), blk, 0, stream>>>(W2, W2t_bf, HID, DD);
  zero_f32<<<dim3(2048), blk, 0, stream>>>((float*)d_ws, zero_floats);

  // GEMM1: feat = x @ W  (bf16 WMMA, f32 out)
  wmma_gemm<0><<<dim3(DD / 64, (Mi + 63) / 64), gblk, 0, stream>>>(
      x_bf, Wt_bf, feat, nullptr, Mi, DD, DD);

  // per-node attention logits
  long long NHt = Nn * HH;
  node_logits_k<<<dim3((unsigned)((NHt + 255) / 256)), blk, 0, stream>>>(
      feat, attn_l, attn_r, el, er, NHt);

  // edge softmax + aggregate (3 passes; destinations live in L2)
  long long EHt = E * HH;
  dim3 egrid((unsigned)((EHt + 255) / 256));
  edge_max_k<<<egrid, blk, 0, stream>>>(src, dst, el, er, menc, EHt);
  edge_sum_k<<<egrid, blk, 0, stream>>>(src, dst, el, er, menc, denom, EHt);
  edge_agg_k<<<egrid, blk, 0, stream>>>(src, dst, el, er, menc, denom, feat, agg, EHt);

  // BN1 -> bf16 (feeds GEMM2)
  const int rowsPer = 256;
  dim3 sgrid((Mi + rowsPer - 1) / rowsPer);
  bn_stats_k<<<sgrid, dim3(128), 0, stream>>>(agg, sum1, sq1, Mi, rowsPer);
  bn_apply_k<true><<<dim3((unsigned)((nx + 255) / 256)), blk, 0, stream>>>(
      agg, sum1, sq1, bn1_g, bn1_b, hn_bf, Mi);

  // GEMM2: act = relu(hn @ W1 + b1) -> bf16
  wmma_gemm<1><<<dim3(HID / 64, (Mi + 63) / 64), gblk, 0, stream>>>(
      hn_bf, W1t_bf, act_bf, b1, Mi, DD, HID);

  // GEMM3: y = act @ W2 + b2 -> f32
  wmma_gemm<2><<<dim3(DD / 64, (Mi + 63) / 64), gblk, 0, stream>>>(
      act_bf, W2t_bf, y, b2, Mi, HID, DD);

  // BN2 -> d_out (f32)
  bn_stats_k<<<sgrid, dim3(128), 0, stream>>>(y, sum2, sq2, Mi, rowsPer);
  bn_apply_k<false><<<dim3((unsigned)((nx + 255) / 256)), blk, 0, stream>>>(
      y, sum2, sq2, bn2_g, bn2_b, d_out, Mi);
}
